// MultiheadRoPESelfAttention_26723286516124
// MI455X (gfx1250) — compile-verified, matrix operand order checked by reading, not run on hardware
//
#include <hip/hip_runtime.h>
#include <hip/hip_bf16.h>

// ---------------------------------------------------------------------------
// MultiheadRoPESelfAttention for MI455X (gfx1250), bf16 WMMA pipeline.
// B=4, S=2048, D=1024, H=16, HDIM=64.
// ---------------------------------------------------------------------------

typedef __attribute__((ext_vector_type(16))) __bf16 v16bf;
typedef __attribute__((ext_vector_type(8)))  __bf16 bf16x8;
typedef __attribute__((ext_vector_type(8)))  float  v8f;

#define GAS __attribute__((address_space(1)))   // global address space
typedef const GAS __bf16* gcbf16p;

#define BATCH   4
#define SEQ     2048
#define DMODEL  1024
#define NHEADS  16
#define HDIM    64
#define NTOK    (BATCH * SEQ)   // 8192

#define ZERO8F  ((v8f){0.f,0.f,0.f,0.f,0.f,0.f,0.f,0.f})

// Load one 16-element bf16 A/B fragment half-slab from GLOBAL memory:
// elements [0..7] from base[0..7] (first K-run), [8..15] from base[16..23]
// (second K-run). Per the CDNA5 16-bit A/B layout, lane<16 covers
// K{0..7,16..23}; lane>=16 covers K{8..15,24..31} when base is pre-offset by
// (lane>=16 ? 8 : 0). Address-space(1) cast forces global_load_b128.
__device__ __forceinline__ v16bf load_frag_g(const __bf16* base) {
    gcbf16p gb = (gcbf16p)base;
    bf16x8 lo = *(const GAS bf16x8*)(gb);
    bf16x8 hi = *(const GAS bf16x8*)(gb + 16);
    v16bf r;
#pragma unroll
    for (int i = 0; i < 8; ++i) { r[i] = lo[i]; r[i + 8] = hi[i]; }
    return r;
}

// Same, from generic (used for LDS-staged data; compiler selects ds_load).
__device__ __forceinline__ v16bf load_frag(const __bf16* base) {
    bf16x8 lo = *(const bf16x8*)(base);
    bf16x8 hi = *(const bf16x8*)(base + 16);
    v16bf r;
#pragma unroll
    for (int i = 0; i < 8; ++i) { r[i] = lo[i]; r[i + 8] = hi[i]; }
    return r;
}

__device__ __forceinline__ void store_bf16_g(__bf16* p, __bf16 v) {
    *(GAS __bf16*)p = v;
}
__device__ __forceinline__ void store_f32_g(float* p, float v) {
    *(GAS float*)p = v;
}

// One wave computes a 64(M) x 64(N) f32 tile of C = A(MxK) * B(NxK)^T.
// 16 accumulators (128 VGPRs) + 8 fragments (64 VGPRs); __launch_bounds__(32,1)
// on callers keeps this spill-free.
__device__ __forceinline__ void wave_gemm_64x64(
    const __bf16* __restrict__ A, int lda,
    const __bf16* __restrict__ Bw, int ldb,
    int K, int m0, int n0, v8f acc[4][4]) {
    const int lane = threadIdx.x & 31;
    const int half = lane >> 4;   // 0: K-runs {0..7,16..23}; 1: {8..15,24..31}
    const int l16  = lane & 15;

#pragma unroll
    for (int i = 0; i < 4; ++i)
#pragma unroll
        for (int j = 0; j < 4; ++j) acc[i][j] = ZERO8F;

    // Hoisted per-lane global pointers; advance 32 elements (64B) per K step.
    gcbf16p pa[4];
    gcbf16p pb[4];
#pragma unroll
    for (int i = 0; i < 4; ++i)
        pa[i] = (gcbf16p)(A + (size_t)(m0 + 16 * i + l16) * lda + half * 8);
#pragma unroll
    for (int j = 0; j < 4; ++j)
        pb[j] = (gcbf16p)(Bw + (size_t)(n0 + 16 * j + l16) * ldb + half * 8);

    for (int k0 = 0; k0 < K; k0 += 32) {
        v16bf af[4], bfm[4];
#pragma unroll
        for (int i = 0; i < 4; ++i) {
            bf16x8 lo = *(const GAS bf16x8*)(pa[i]);
            bf16x8 hi = *(const GAS bf16x8*)(pa[i] + 16);
#pragma unroll
            for (int q = 0; q < 8; ++q) { af[i][q] = lo[q]; af[i][q + 8] = hi[q]; }
            __builtin_prefetch(pa[i] + 32, 0, 3);   // global_prefetch_b8
            pa[i] += 32;
        }
#pragma unroll
        for (int j = 0; j < 4; ++j) {
            bf16x8 lo = *(const GAS bf16x8*)(pb[j]);
            bf16x8 hi = *(const GAS bf16x8*)(pb[j] + 16);
#pragma unroll
            for (int q = 0; q < 8; ++q) { bfm[j][q] = lo[q]; bfm[j][q + 8] = hi[q]; }
            __builtin_prefetch(pb[j] + 32, 0, 3);
            pb[j] += 32;
        }
#pragma unroll
        for (int i = 0; i < 4; ++i)
#pragma unroll
            for (int j = 0; j < 4; ++j)
                acc[i][j] = __builtin_amdgcn_wmma_f32_16x16x32_bf16(
                    false, af[i], false, bfm[j], (short)0, acc[i][j], false, false);
    }
}

// ---------------------------------------------------------------------------
// Kernel 0: fp32 -> bf16 conversion (grid-stride)
// ---------------------------------------------------------------------------
__global__ void __launch_bounds__(256) f32_to_bf16_kernel(
    const float* __restrict__ src, __bf16* __restrict__ dst, int n) {
    int i = blockIdx.x * blockDim.x + threadIdx.x;
    int stride = gridDim.x * blockDim.x;
    for (; i < n; i += stride) {
        float v = *(const GAS float*)(src + i);
        store_bf16_g(dst + i, (__bf16)v);
    }
}

// ---------------------------------------------------------------------------
// Kernel 1: QKV GEMM (8192x1024 @ 1024x3072) + fused RoPE + head packing.
//   Q,K -> [B,H,S,HDIM] row-major (contiguous HDIM)      (A/B frags for QK^T)
//   V   -> [B,H,HDIM,S] transposed (contiguous S)        (B frag for P@V)
// RoPE pairs (even,odd feature dims) live in adjacent lanes of the C tile,
// so the partner value is fetched with __shfl_xor(.,1).
// ---------------------------------------------------------------------------
__global__ void __launch_bounds__(32, 1) qkv_rope_pack_kernel(
    const __bf16* __restrict__ Xb,
    const __bf16* __restrict__ Wqb,
    const int* __restrict__ tpos,
    __bf16* __restrict__ Qh,
    __bf16* __restrict__ Kh,
    __bf16* __restrict__ Vt) {
    v8f acc[4][4];
    const int m0 = blockIdx.x * 64;   // token rows
    const int n0 = blockIdx.y * 64;   // qkv features
    wave_gemm_64x64(Xb, DMODEL, Wqb, DMODEL, DMODEL, m0, n0, acc);

    const int lane = threadIdx.x & 31;
    const int half = lane >> 4;
    const int l16  = lane & 15;

#pragma unroll
    for (int i = 0; i < 4; ++i) {
#pragma unroll
        for (int j = 0; j < 4; ++j) {
            const int f    = n0 + 16 * j + l16;   // feature in [0, 3072)
            const int kind = f >> 10;             // 0:q 1:k 2:v
            const int d    = f & (DMODEL - 1);
            const int h    = d >> 6;
            const int dh   = d & (HDIM - 1);
            // RoPE frequency depends only on the feature column -> hoist.
            const float freq =
                __powf(10000.f, -(float)(dh & ~1) / (float)HDIM);
#pragma unroll
            for (int e = 0; e < 8; ++e) {
                const int n = m0 + 16 * i + e + half * 8;   // token index
                const int b = n >> 11;
                const int s = n & (SEQ - 1);
                const float v  = acc[i][j][e];
                const float rp = __shfl_xor(v, 1);          // pair partner
                if (kind == 2) {
                    store_bf16_g(
                        &Vt[(((size_t)(b * NHEADS + h)) * HDIM + dh) * SEQ + s],
                        (__bf16)v);
                } else {
                    const int pos = *(const GAS int*)(tpos + s);
                    float sn, cs;
                    __sincosf((float)pos * freq, &sn, &cs);
                    const float outv = (f & 1) ? (rp * sn + v * cs)
                                               : (v * cs - rp * sn);
                    const size_t idx =
                        (((size_t)(b * NHEADS + h)) * SEQ + s) * HDIM + dh;
                    if (kind == 0) store_bf16_g(&Qh[idx], (__bf16)outv);
                    else           store_bf16_g(&Kh[idx], (__bf16)outv);
                }
            }
        }
    }
}

// ---------------------------------------------------------------------------
// Kernel 2: causal flash attention. One wave (32 thr) per 16-query tile per
// (b,h). 32-key steps: 4 score WMMAs + online softmax (16-lane shuffle
// reductions) + LDS C->A transpose of exp(P) + 4 P@V WMMAs.
// ---------------------------------------------------------------------------
__global__ void __launch_bounds__(32, 1) flash_attn_kernel(
    const __bf16* __restrict__ Qh,
    const __bf16* __restrict__ Kh,
    const __bf16* __restrict__ Vt,
    __bf16* __restrict__ Ob) {
    __shared__ __align__(16) __bf16 Plds[16 * 40];  // 16 rows x 32 keys, pad to 40

    const int qt = blockIdx.x;           // query tile 0..127
    const int bh = blockIdx.y;           // 0..63
    const int b  = bh >> 4, h = bh & 15;
    const int lane = threadIdx.x & 31;
    const int half = lane >> 4;
    const int l16  = lane & 15;

    const __bf16* Q  = Qh + (size_t)bh * SEQ * HDIM;
    const __bf16* Kp = Kh + (size_t)bh * SEQ * HDIM;
    const __bf16* Vp = Vt + (size_t)bh * HDIM * SEQ;

    // Q fragments: row = qt*16 + l16, two K-chunks (dh 0..31 / 32..63)
    v16bf qa[2];
#pragma unroll
    for (int c = 0; c < 2; ++c)
        qa[c] = load_frag_g(Q + (size_t)(qt * 16 + l16) * HDIM + c * 32 + half * 8);

    v8f o[4];
#pragma unroll
    for (int t = 0; t < 4; ++t) o[t] = ZERO8F;
    float m_run[8], l_run[8];
#pragma unroll
    for (int e = 0; e < 8; ++e) { m_run[e] = -1e30f; l_run[e] = 0.f; }

    const int kend = qt * 16 + 16;
    for (int kb = 0; kb < kend; kb += 32) {
        // ---- scores: two 16-key C tiles, K=64 via two chained WMMAs each
        v8f sc[2];
#pragma unroll
        for (int j = 0; j < 2; ++j) {
            const int key = kb + 16 * j + l16;
            v16bf kf0 = load_frag_g(Kp + (size_t)key * HDIM +  0 + half * 8);
            v16bf kf1 = load_frag_g(Kp + (size_t)key * HDIM + 32 + half * 8);
            v8f z = ZERO8F;
            z = __builtin_amdgcn_wmma_f32_16x16x32_bf16(false, qa[0], false, kf0,
                                                        (short)0, z, false, false);
            z = __builtin_amdgcn_wmma_f32_16x16x32_bf16(false, qa[1], false, kf1,
                                                        (short)0, z, false, false);
            sc[j] = z;
        }
        // ---- scale + causal mask + per-row tile max
        float tm[8];
#pragma unroll
        for (int e = 0; e < 8; ++e) {
            const int row = qt * 16 + e + half * 8;
#pragma unroll
            for (int j = 0; j < 2; ++j) {
                const int key = kb + 16 * j + l16;
                float v = sc[j][e] * 0.125f;                 // 1/sqrt(64)
                if (key > row) v = -1e30f;
                sc[j][e] = v;
            }
            tm[e] = fmaxf(sc[0][e], sc[1][e]);
        }
#pragma unroll
        for (int off = 1; off < 16; off <<= 1)
#pragma unroll
            for (int e = 0; e < 8; ++e)
                tm[e] = fmaxf(tm[e], __shfl_xor(tm[e], off));
        // ---- online softmax update; write exp(P) into LDS (bf16)
        float alpha[8], ps[8];
#pragma unroll
        for (int e = 0; e < 8; ++e) {
            const float mn = fmaxf(m_run[e], tm[e]);
            alpha[e] = __expf(m_run[e] - mn);
            m_run[e] = mn;
            const float p0 = __expf(sc[0][e] - mn);
            const float p1 = __expf(sc[1][e] - mn);
            ps[e] = p0 + p1;
            const int row = e + half * 8;
            Plds[row * 40 +      l16] = (__bf16)p0;
            Plds[row * 40 + 16 + l16] = (__bf16)p1;
        }
#pragma unroll
        for (int off = 1; off < 16; off <<= 1)
#pragma unroll
            for (int e = 0; e < 8; ++e)
                ps[e] += __shfl_xor(ps[e], off);
#pragma unroll
        for (int e = 0; e < 8; ++e) l_run[e] = l_run[e] * alpha[e] + ps[e];
#pragma unroll
        for (int t = 0; t < 4; ++t)
#pragma unroll
            for (int e = 0; e < 8; ++e) o[t][e] *= alpha[e];

        __syncthreads();
        // ---- P (16x32) as A-fragment from LDS; V^T B-fragments; 4 WMMAs
        const v16bf pa = load_frag(&Plds[l16 * 40 + half * 8]);
#pragma unroll
        for (int t = 0; t < 4; ++t) {
            const int dh = t * 16 + l16;
            const v16bf vf = load_frag_g(Vp + (size_t)dh * SEQ + kb + half * 8);
            o[t] = __builtin_amdgcn_wmma_f32_16x16x32_bf16(
                false, pa, false, vf, (short)0, o[t], false, false);
        }
        __syncthreads();
    }

    // ---- normalize and pack output as [n, d] = [b*S+s, h*64+dh] bf16
#pragma unroll
    for (int t = 0; t < 4; ++t)
#pragma unroll
        for (int e = 0; e < 8; ++e) {
            const int s = qt * 16 + e + half * 8;
            const size_t n = (size_t)b * SEQ + s;
            store_bf16_g(&Ob[n * DMODEL + h * HDIM + t * 16 + l16],
                         (__bf16)(o[t][e] / l_run[e]));
        }
}

// ---------------------------------------------------------------------------
// Kernel 3: output projection, fp32 result to d_out
// ---------------------------------------------------------------------------
__global__ void __launch_bounds__(32, 1) out_proj_kernel(
    const __bf16* __restrict__ Ob,
    const __bf16* __restrict__ Wob,
    float* __restrict__ out) {
    v8f acc[4][4];
    const int m0 = blockIdx.x * 64;
    const int n0 = blockIdx.y * 64;
    wave_gemm_64x64(Ob, DMODEL, Wob, DMODEL, DMODEL, m0, n0, acc);

    const int lane = threadIdx.x & 31;
    const int half = lane >> 4;
    const int l16  = lane & 15;
#pragma unroll
    for (int i = 0; i < 4; ++i)
#pragma unroll
        for (int j = 0; j < 4; ++j)
#pragma unroll
            for (int e = 0; e < 8; ++e) {
                const size_t r = (size_t)(m0 + 16 * i + e + half * 8);
                store_f32_g(&out[r * DMODEL + n0 + 16 * j + l16], acc[i][j][e]);
            }
}

// ---------------------------------------------------------------------------
extern "C" void kernel_launch(void* const* d_in, const int* in_sizes, int n_in,
                              void* d_out, int out_size, void* d_ws, size_t ws_size,
                              hipStream_t stream) {
    (void)in_sizes; (void)n_in; (void)out_size; (void)ws_size;
    const float* x    = (const float*)d_in[0];
    const int*   tpos = (const int*)d_in[1];
    const float* Wqkv = (const float*)d_in[2];
    const float* Wout = (const float*)d_in[3];
    float* out = (float*)d_out;

    // Workspace partition (all 16B-aligned slabs)
    char* ws = (char*)d_ws;
    __bf16* Xb  = (__bf16*)ws; ws += (size_t)NTOK * DMODEL * 2;          // 16 MiB
    __bf16* Wqb = (__bf16*)ws; ws += (size_t)3 * DMODEL * DMODEL * 2;    //  6 MiB
    __bf16* Wob = (__bf16*)ws; ws += (size_t)DMODEL * DMODEL * 2;        //  2 MiB
    __bf16* Qh  = (__bf16*)ws; ws += (size_t)NTOK * DMODEL * 2;          // 16 MiB
    __bf16* Kh  = (__bf16*)ws; ws += (size_t)NTOK * DMODEL * 2;          // 16 MiB
    __bf16* Vt  = (__bf16*)ws; ws += (size_t)NTOK * DMODEL * 2;          // 16 MiB
    __bf16* Ob  = (__bf16*)ws; ws += (size_t)NTOK * DMODEL * 2;          // 16 MiB

    f32_to_bf16_kernel<<<1024, 256, 0, stream>>>(x,    Xb,  NTOK * DMODEL);
    f32_to_bf16_kernel<<<1024, 256, 0, stream>>>(Wqkv, Wqb, 3 * DMODEL * DMODEL);
    f32_to_bf16_kernel<<<1024, 256, 0, stream>>>(Wout, Wob, DMODEL * DMODEL);

    dim3 gQKV(NTOK / 64, (3 * DMODEL) / 64);          // 128 x 48 waves
    qkv_rope_pack_kernel<<<gQKV, 32, 0, stream>>>(Xb, Wqb, tpos, Qh, Kh, Vt);

    dim3 gAtt(SEQ / 16, BATCH * NHEADS);              // 128 x 64 waves
    flash_attn_kernel<<<gAtt, 32, 0, stream>>>(Qh, Kh, Vt, Ob);

    dim3 gOut(NTOK / 64, DMODEL / 64);                // 128 x 16 waves
    out_proj_kernel<<<gOut, 32, 0, stream>>>(Ob, Wob, out);
}
